// KeypointGenerator_15951508537874
// MI455X (gfx1250) — matrix operand order repaired and checked by reading, not compiled, not run on hardware
//
#include <hip/hip_runtime.h>

// ---------------------------------------------------------------------------
// Types for CDNA5 WMMA (wave32): V_WMMA_F32_16X16X32_F16
// ---------------------------------------------------------------------------
typedef __attribute__((ext_vector_type(16))) _Float16 v16h;
typedef __attribute__((ext_vector_type(8)))  _Float16 v8h;
typedef __attribute__((ext_vector_type(8)))  float    v8f;

#define NUM_PT 8192
#define TOPK   32
#define HW     19200

__device__ __forceinline__ float lrelu(float x) { return x > 0.f ? x : 0.01f * x; }

__device__ __forceinline__ v16h make16(v8h lo, v8h hi) {
  return __builtin_shufflevector(lo, hi, 0,1,2,3,4,5,6,7,8,9,10,11,12,13,14,15);
}

// A-matrix fragment (16x32 f16). ISA layout: lane L -> row M = L&15;
// within K-chunk: a[0..7] = K {h*8 .. h*8+7}, a[8..15] = K {16+h*8 ..}, h = L>>4.
__device__ __forceinline__ v16h load_a_frag(const _Float16* rowp, int kbase, int lane) {
  const int h8 = ((lane >> 4) & 1) * 8;
  v8h lo = *(const v8h*)(rowp + kbase + h8);
  v8h hi = *(const v8h*)(rowp + kbase + 16 + h8);
  return make16(lo, hi);
}

// B-matrix fragment (32x16 f16) from row-major W[out][in]:
// lane L -> col N = L&15; b[0..7] = K {h*8 .. h*8+7}, b[8..15] = K {16+h*8 ..}, h = L>>4.
__device__ __forceinline__ v16h load_b_frag(const _Float16* W, int kdim, int nbase,
                                            int kbase, int lane) {
  const _Float16* p = W + (size_t)(nbase + (lane & 15)) * kdim + kbase + ((lane >> 4) & 1) * 8;
  v8h lo = *(const v8h*)p;
  v8h hi = *(const v8h*)(p + 16);
  return make16(lo, hi);
}

__device__ __forceinline__ v8f wmma16(v16h a, v16h b, v8f c) {
  // (neg_a, A, neg_b, B, c_mod, C, reuse_a, reuse_b)
  return __builtin_amdgcn_wmma_f32_16x16x32_f16(false, a, false, b, (short)0, c, false, false);
}

// ---------------------------------------------------------------------------
// K0: f32 -> f16 conversion (weights)
// ---------------------------------------------------------------------------
__global__ void f2h_kernel(const float* __restrict__ src, _Float16* __restrict__ dst, int n) {
  int i = blockIdx.x * 256 + threadIdx.x;
  if (i < n) dst[i] = (_Float16)src[i];
}

// ---------------------------------------------------------------------------
// K1: gather per-point image features (raw f16 + lrelu'd f16)
// ---------------------------------------------------------------------------
__global__ __launch_bounds__(256) void gather_feat_kernel(
    const float* __restrict__ img, const int* __restrict__ choose,
    _Float16* __restrict__ feat_h, _Float16* __restrict__ feat_lr) {
  int n = blockIdx.x * 256 + threadIdx.x;
  int pix = choose[n];
  #pragma unroll
  for (int c = 0; c < 32; ++c) {
    float v = img[c * HW + pix];
    feat_h[n * 32 + c]  = (_Float16)v;
    feat_lr[n * 32 + c] = (_Float16)lrelu(v);
  }
}

// ---------------------------------------------------------------------------
// K2: per-point pf MLP  c[3] -> 64 (lrelu) -> 128  (VALU; tiny)
// ---------------------------------------------------------------------------
__global__ __launch_bounds__(256) void pf_kernel(
    const float* __restrict__ cloud,
    const float* __restrict__ w1, const float* __restrict__ b1,
    const float* __restrict__ w2, const float* __restrict__ b2,
    _Float16* __restrict__ pf_h, _Float16* __restrict__ pf_lr) {
  int n = blockIdx.x * 256 + threadIdx.x;
  float x0 = cloud[n * 3 + 0], x1 = cloud[n * 3 + 1], x2 = cloud[n * 3 + 2];
  float h[64];
  #pragma unroll
  for (int o = 0; o < 64; ++o) {
    float a = b1[o] + w1[o * 3 + 0] * x0 + w1[o * 3 + 1] * x1 + w1[o * 3 + 2] * x2;
    h[o] = lrelu(a);
  }
  for (int o = 0; o < 128; ++o) {
    float a = b2[o];
    const float* wo = w2 + o * 64;
    #pragma unroll
    for (int c = 0; c < 64; ++c) a += wo[c] * h[c];
    pf_h[n * 128 + o]  = (_Float16)a;
    pf_lr[n * 128 + o] = (_Float16)lrelu(a);
  }
}

// ---------------------------------------------------------------------------
// K3: KNN (top-32 incl self) + softmax(-dist) weights.
// ---------------------------------------------------------------------------
__global__ __launch_bounds__(256) void knn_kernel(
    const float* __restrict__ cloud, int* __restrict__ idxArr, float* __restrict__ wgtArr) {
  __shared__ float sx[256], sy[256], sz[256];
  int i = blockIdx.x * 256 + threadIdx.x;
  float cx = cloud[i * 3 + 0], cy = cloud[i * 3 + 1], cz = cloud[i * 3 + 2];
  float dist[TOPK];
  int   idr[TOPK];
  #pragma unroll
  for (int k = 0; k < TOPK; ++k) { dist[k] = 3.0e38f; idr[k] = 0; }

  for (int tile = 0; tile < NUM_PT; tile += 256) {
    int j = tile + threadIdx.x;
    sx[threadIdx.x] = cloud[j * 3 + 0];
    sy[threadIdx.x] = cloud[j * 3 + 1];
    sz[threadIdx.x] = cloud[j * 3 + 2];
    __syncthreads();
    for (int u = 0; u < 256; ++u) {
      float dx = cx - sx[u], dy = cy - sy[u], dz = cz - sz[u];
      float d2 = dx * dx + dy * dy + dz * dz;
      if (d2 < dist[TOPK - 1]) {
        dist[TOPK - 1] = d2;
        idr[TOPK - 1]  = tile + u;
        #pragma unroll
        for (int k = TOPK - 1; k > 0; --k) {
          float a_ = dist[k - 1], b_ = dist[k];
          int   ia = idr[k - 1],  ib = idr[k];
          bool sw = b_ < a_;
          dist[k - 1] = sw ? b_ : a_;
          dist[k]     = sw ? a_ : b_;
          idr[k - 1]  = sw ? ib : ia;
          idr[k]      = sw ? ia : ib;
        }
      }
    }
    __syncthreads();
  }
  float m = -sqrtf(dist[0]);
  float s = 0.f;
  #pragma unroll
  for (int k = 0; k < TOPK; ++k) {
    float e = expf(-sqrtf(dist[k]) - m);
    dist[k] = e;
    s += e;
  }
  float inv = 1.f / s;
  #pragma unroll
  for (int k = 0; k < TOPK; ++k) {
    idxArr[i * TOPK + k] = idr[k];
    wgtArr[i * TOPK + k] = dist[k] * inv;
  }
}

// ---------------------------------------------------------------------------
// K4: fused per-point neighbor MLPs with WMMA.
// One wave per point (8 points / 256-thread block).
// ALL weights staged in LDS; B fragments are ds-loads that cannot be
// hoisted/CSE'd across the __syncthreads() fences -> no scratch spills.
// LDS: 64K (W1) + 64K (W2) + 4K (C1) + 16K (C2) + 64K (H) = 212 KB < 320 KB.
// ---------------------------------------------------------------------------
__global__ __launch_bounds__(256) void point_mlp_kernel(
    const _Float16* __restrict__ feat_h, const _Float16* __restrict__ pf_h,
    const int* __restrict__ idxArr, const float* __restrict__ wgtArr,
    const _Float16* __restrict__ c1h, const float* __restrict__ c1b,
    const _Float16* __restrict__ c2h, const float* __restrict__ c2b,
    const _Float16* __restrict__ p1h, const float* __restrict__ p1b,
    const _Float16* __restrict__ p2h, const float* __restrict__ p2b,
    _Float16* __restrict__ fpool_lr, _Float16* __restrict__ fppool_lr) {
  __shared__ __align__(16) _Float16 sW1[256 * 128];   // psconv1 [256][128]
  __shared__ __align__(16) _Float16 sW2[128 * 256];   // psconv2 [128][256]
  __shared__ __align__(16) _Float16 sC1[64 * 32];     // conv1   [64][32]
  __shared__ __align__(16) _Float16 sC2[128 * 64];    // conv2   [128][64]
  __shared__ __align__(16) _Float16 sH[8][16 * 256];  // per-wave hidden staging

  const int tid = threadIdx.x;
  // cooperative weight staging (v8h = 16B chunks)
  {
    const v8h* s1 = (const v8h*)p1h; v8h* d1 = (v8h*)sW1;
    for (int i = tid; i < 4096; i += 256) d1[i] = s1[i];
    const v8h* s2 = (const v8h*)p2h; v8h* d2 = (v8h*)sW2;
    for (int i = tid; i < 4096; i += 256) d2[i] = s2[i];
    const v8h* s3 = (const v8h*)c1h; v8h* d3 = (v8h*)sC1;
    for (int i = tid; i < 256; i += 256) d3[i] = s3[i];
    const v8h* s4 = (const v8h*)c2h; v8h* d4 = (v8h*)sC2;
    for (int i = tid; i < 1024; i += 256) d4[i] = s4[i];
  }
  __syncthreads();

  const int lane = tid & 31;
  const int wave = tid >> 5;
  const int n    = blockIdx.x * 8 + wave;
  const int mr   = lane & 15;   // A row / C column within tile
  const int hs   = lane >> 4;   // half select
  _Float16* H = sH[wave];
  const v8f vzero = {0.f, 0.f, 0.f, 0.f, 0.f, 0.f, 0.f, 0.f};

  // ============================ fp path ============================
  float cmax[8];
  #pragma unroll
  for (int q = 0; q < 8; ++q) cmax[q] = -3.0e38f;

  #pragma unroll 1
  for (int t = 0; t < 2; ++t) {
    const int j = idxArr[n * 32 + t * 16 + mr];
    const _Float16* arow = pf_h + (size_t)j * 128;

    v8f acc[16];
    #pragma unroll
    for (int q = 0; q < 16; ++q) acc[q] = vzero;
    #pragma unroll
    for (int kk = 0; kk < 4; ++kk) {
      v16h a = load_a_frag(arow, kk * 32, lane);
      #pragma unroll
      for (int nt = 0; nt < 16; ++nt) {
        v16h b = load_b_frag(sW1, 128, nt * 16, kk * 32, lane);
        acc[nt] = wmma16(a, b, acc[nt]);
      }
    }
    // bias + lrelu -> LDS hidden [16][256]
    #pragma unroll
    for (int nt = 0; nt < 16; ++nt) {
      const int Nc = nt * 16 + mr;
      const float bb = p1b[Nc];
      #pragma unroll
      for (int r = 0; r < 8; ++r) {
        float v = lrelu(acc[nt][r] + bb);
        H[(hs * 8 + r) * 256 + Nc] = (_Float16)v;
      }
    }
    __syncthreads();

    v8f acc2[8];
    #pragma unroll
    for (int q = 0; q < 8; ++q) acc2[q] = vzero;
    #pragma unroll
    for (int kk = 0; kk < 8; ++kk) {
      v16h a = load_a_frag(H + mr * 256, kk * 32, lane);
      #pragma unroll
      for (int nt = 0; nt < 8; ++nt) {
        v16h b = load_b_frag(sW2, 256, nt * 16, kk * 32, lane);
        acc2[nt] = wmma16(a, b, acc2[nt]);
      }
    }
    float wg[8];
    #pragma unroll
    for (int r = 0; r < 8; ++r) wg[r] = wgtArr[n * 32 + t * 16 + hs * 8 + r];
    #pragma unroll
    for (int nt = 0; nt < 8; ++nt) {
      const int Nc = nt * 16 + mr;
      const float bb = p2b[Nc];
      #pragma unroll
      for (int r = 0; r < 8; ++r)
        cmax[nt] = fmaxf(cmax[nt], (acc2[nt][r] + bb) * wg[r]);
    }
    __syncthreads();
  }
  #pragma unroll
  for (int nt = 0; nt < 8; ++nt) {
    float o = __shfl_xor(cmax[nt], 16, 32);
    float m = fmaxf(cmax[nt], o);
    if (lane < 16) fppool_lr[n * 128 + nt * 16 + lane] = (_Float16)lrelu(m);
  }

  // ============================ f path ============================
  #pragma unroll
  for (int q = 0; q < 8; ++q) cmax[q] = -3.0e38f;

  #pragma unroll 1
  for (int t = 0; t < 2; ++t) {
    const int j = idxArr[n * 32 + t * 16 + mr];
    const _Float16* arow = feat_h + (size_t)j * 32;

    v8f accf[4];
    #pragma unroll
    for (int q = 0; q < 4; ++q) accf[q] = vzero;
    {
      v16h a = load_a_frag(arow, 0, lane);
      #pragma unroll
      for (int nt = 0; nt < 4; ++nt) {
        v16h b = load_b_frag(sC1, 32, nt * 16, 0, lane);
        accf[nt] = wmma16(a, b, accf[nt]);
      }
    }
    #pragma unroll
    for (int nt = 0; nt < 4; ++nt) {
      const int Nc = nt * 16 + mr;
      const float bb = c1b[Nc];
      #pragma unroll
      for (int r = 0; r < 8; ++r) {
        float v = lrelu(accf[nt][r] + bb);
        H[(hs * 8 + r) * 64 + Nc] = (_Float16)v;
      }
    }
    __syncthreads();

    v8f acc2[8];
    #pragma unroll
    for (int q = 0; q < 8; ++q) acc2[q] = vzero;
    #pragma unroll
    for (int kk = 0; kk < 2; ++kk) {
      v16h a = load_a_frag(H + mr * 64, kk * 32, lane);
      #pragma unroll
      for (int nt = 0; nt < 8; ++nt) {
        v16h b = load_b_frag(sC2, 64, nt * 16, kk * 32, lane);
        acc2[nt] = wmma16(a, b, acc2[nt]);
      }
    }
    float wg[8];
    #pragma unroll
    for (int r = 0; r < 8; ++r) wg[r] = wgtArr[n * 32 + t * 16 + hs * 8 + r];
    #pragma unroll
    for (int nt = 0; nt < 8; ++nt) {
      const int Nc = nt * 16 + mr;
      const float bb = c2b[Nc];
      #pragma unroll
      for (int r = 0; r < 8; ++r)
        cmax[nt] = fmaxf(cmax[nt], (acc2[nt][r] + bb) * wg[r]);
    }
    __syncthreads();
  }
  #pragma unroll
  for (int nt = 0; nt < 8; ++nt) {
    float o = __shfl_xor(cmax[nt], 16, 32);
    float m = fmaxf(cmax[nt], o);
    if (lane < 16) fpool_lr[n * 128 + nt * 16 + lane] = (_Float16)lrelu(m);
  }
}

// ---------------------------------------------------------------------------
// K5: fused = lrelu(concat[fp|feat|f|pf]) @ final_w.T + b   (WMMA GEMM 416->128)
// final_w staged in LDS (104 KB) to keep B fragments as short-lived ds loads.
// ---------------------------------------------------------------------------
__global__ __launch_bounds__(256) void final_gemm_kernel(
    const _Float16* __restrict__ fppool_lr, const _Float16* __restrict__ feat_lr,
    const _Float16* __restrict__ fpool_lr,  const _Float16* __restrict__ pf_lr,
    const _Float16* __restrict__ fwh, const float* __restrict__ fb,
    float* __restrict__ fused) {
  __shared__ __align__(16) _Float16 sFW[128 * 416];
  const int tid = threadIdx.x;
  {
    const v8h* s = (const v8h*)fwh; v8h* d = (v8h*)sFW;
    for (int i = tid; i < (128 * 416) / 8; i += 256) d[i] = s[i];
  }
  __syncthreads();

  const int lane = tid & 31;
  const int wave = tid >> 5;
  const int mt   = blockIdx.x * 8 + wave;   // 512 M-tiles
  const int mr   = lane & 15;
  const int hs   = lane >> 4;
  const int row  = mt * 16 + mr;
  const v8f vzero = {0.f, 0.f, 0.f, 0.f, 0.f, 0.f, 0.f, 0.f};
  v8f acc[8];
  #pragma unroll
  for (int q = 0; q < 8; ++q) acc[q] = vzero;

  #pragma unroll
  for (int kk = 0; kk < 13; ++kk) {
    const _Float16* src;
    int cb;
    if (kk < 4)      { src = fppool_lr + (size_t)row * 128; cb = kk * 32; }
    else if (kk == 4){ src = feat_lr   + (size_t)row * 32;  cb = 0; }
    else if (kk < 9) { src = fpool_lr  + (size_t)row * 128; cb = (kk - 5) * 32; }
    else             { src = pf_lr     + (size_t)row * 128; cb = (kk - 9) * 32; }
    v16h a = load_a_frag(src, cb, lane);
    #pragma unroll
    for (int nt = 0; nt < 8; ++nt) {
      v16h b = load_b_frag(sFW, 416, nt * 16, kk * 32, lane);
      acc[nt] = wmma16(a, b, acc[nt]);
    }
  }
  #pragma unroll
  for (int nt = 0; nt < 8; ++nt) {
    const int Nc = nt * 16 + mr;
    const float bb = fb[Nc];
    #pragma unroll
    for (int r = 0; r < 8; ++r) {
      const int M = mt * 16 + hs * 8 + r;
      fused[(size_t)M * 128 + Nc] = acc[nt][r] + bb;
    }
  }
}

// ---------------------------------------------------------------------------
// K6: seg + dis heads (conv3 128->64, 1x1 64->32, 1x1 32->1), LDS row tile.
// ---------------------------------------------------------------------------
__global__ __launch_bounds__(64) void heads_kernel(
    const float* __restrict__ fused,
    const float* __restrict__ s1w, const float* __restrict__ s1b,
    const float* __restrict__ s2w, const float* __restrict__ s2b,
    const float* __restrict__ s3w, const float* __restrict__ s3b,
    const float* __restrict__ d1w, const float* __restrict__ d1b,
    const float* __restrict__ d2w, const float* __restrict__ d2b,
    const float* __restrict__ d3w, const float* __restrict__ d3b,
    float* __restrict__ out) {
  __shared__ float sF[66 * 128];   // rows n0-1 .. n0+64 (halo)
  const int n0 = blockIdx.x * 64;
  for (int i = threadIdx.x; i < 66 * 128; i += 64) {
    int rr = i >> 7, c = i & 127;
    int ns = n0 - 1 + rr;
    sF[i] = (ns >= 0 && ns < NUM_PT) ? fused[(size_t)ns * 128 + c] : 0.f;
  }
  __syncthreads();
  const float* x0 = sF + threadIdx.x * 128;
  const float* x1 = x0 + 128;
  const float* x2 = x1 + 128;
  const int n = n0 + threadIdx.x;
  for (int head = 0; head < 2; ++head) {
    const float* w1 = head ? d1w : s1w;
    const float* b1 = head ? d1b : s1b;
    const float* w2 = head ? d2w : s2w;
    const float* b2 = head ? d2b : s2b;
    const float* w3 = head ? d3w : s3w;
    const float* b3 = head ? d3b : s3b;
    float h[64];
    for (int o = 0; o < 64; ++o) {
      float a = b1[o];
      const float* wo = w1 + o * 384;
      for (int c = 0; c < 128; ++c)
        a += wo[c * 3 + 0] * x0[c] + wo[c * 3 + 1] * x1[c] + wo[c * 3 + 2] * x2[c];
      h[o] = lrelu(a);
    }
    float h2[32];
    for (int o = 0; o < 32; ++o) {
      float a = b2[o];
      const float* wo = w2 + o * 64;
      for (int c = 0; c < 64; ++c) a += wo[c] * h[c];
      h2[o] = lrelu(a);
    }
    float a3 = b3[0];
    for (int c = 0; c < 32; ++c) a3 += w3[c] * h2[c];
    out[(head ? 24 : 16408) + n] = a3;   // dis_pre @24, seg_pre @16408
  }
}

// ---------------------------------------------------------------------------
// K7: dis_center = softmax_n(|cloud + t|)  (single block, LDS reductions)
// ---------------------------------------------------------------------------
__global__ __launch_bounds__(1024) void discenter_kernel(
    const float* __restrict__ cloud, const float* __restrict__ tt,
    float* __restrict__ dcen, float* __restrict__ out) {
  __shared__ float red[1024];
  const int tid = threadIdx.x;
  const float t0 = tt[0], t1 = tt[1], t2 = tt[2];
  float r[8];
  float m = -3.0e38f;
  #pragma unroll
  for (int q = 0; q < 8; ++q) {
    int n = tid + q * 1024;
    float a = cloud[n * 3 + 0] + t0;
    float b = cloud[n * 3 + 1] + t1;
    float c = cloud[n * 3 + 2] + t2;
    r[q] = sqrtf(a * a + b * b + c * c);
    m = fmaxf(m, r[q]);
  }
  red[tid] = m; __syncthreads();
  for (int st = 512; st > 0; st >>= 1) {
    if (tid < st) red[tid] = fmaxf(red[tid], red[tid + st]);
    __syncthreads();
  }
  m = red[0]; __syncthreads();
  float s = 0.f;
  #pragma unroll
  for (int q = 0; q < 8; ++q) { r[q] = expf(r[q] - m); s += r[q]; }
  red[tid] = s; __syncthreads();
  for (int st = 512; st > 0; st >>= 1) {
    if (tid < st) red[tid] += red[tid + st];
    __syncthreads();
  }
  float inv = 1.f / red[0];
  #pragma unroll
  for (int q = 0; q < 8; ++q) {
    int n = tid + q * 1024;
    float v = r[q] * inv;
    dcen[n] = v;
    out[8216 + n] = v;   // dis_center @24+8192
  }
}

// ---------------------------------------------------------------------------
// K8: g[c] = sum_n fused[n][c] * dis_center[n] * seg[n]
// ---------------------------------------------------------------------------
__global__ __launch_bounds__(256) void gpool_kernel(
    const float* __restrict__ fused, const float* __restrict__ dcen,
    const float* __restrict__ seg, float* __restrict__ g) {
  __shared__ float red[256];
  const int c = blockIdx.x;
  float a = 0.f;
  for (int n = threadIdx.x; n < NUM_PT; n += 256)
    a += fused[(size_t)n * 128 + c] * dcen[n] * seg[n];
  red[threadIdx.x] = a; __syncthreads();
  for (int st = 128; st > 0; st >>= 1) {
    if (threadIdx.x < st) red[threadIdx.x] += red[threadIdx.x + st];
    __syncthreads();
  }
  if (threadIdx.x == 0) g[c] = red[0];
}

// ---------------------------------------------------------------------------
// K9: keypoint MLP 128->90->64->24 (single block)
// ---------------------------------------------------------------------------
__global__ __launch_bounds__(128) void kp_kernel(
    const float* __restrict__ g,
    const float* __restrict__ l1w, const float* __restrict__ l1b,
    const float* __restrict__ l2w, const float* __restrict__ l2b,
    const float* __restrict__ l3w, const float* __restrict__ l3b,
    float* __restrict__ out) {
  __shared__ float k1[90];
  __shared__ float k2[64];
  const int tid = threadIdx.x;
  if (tid < 90) {
    float a = l1b[tid];
    for (int c = 0; c < 128; ++c) a += l1w[tid * 128 + c] * g[c];
    k1[tid] = lrelu(a);
  }
  __syncthreads();
  if (tid < 64) {
    float a = l2b[tid];
    for (int c = 0; c < 90; ++c) a += l2w[tid * 90 + c] * k1[c];
    k2[tid] = lrelu(a);
  }
  __syncthreads();
  if (tid < 24) {
    float a = l3b[tid];
    for (int c = 0; c < 64; ++c) a += l3w[tid * 64 + c] * k2[c];
    out[tid] = a;   // keypoints @0
  }
}

// ---------------------------------------------------------------------------
// Host launch
// ---------------------------------------------------------------------------
extern "C" void kernel_launch(void* const* d_in, const int* in_sizes, int n_in,
                              void* d_out, int out_size, void* d_ws, size_t ws_size,
                              hipStream_t stream) {
  (void)in_sizes; (void)n_in; (void)out_size; (void)ws_size;
  const float* seg      = (const float*)d_in[0];
  const float* img      = (const float*)d_in[1];
  const float* cloud    = (const float*)d_in[2];
  const float* t        = (const float*)d_in[3];
  const int*   choose   = (const int*)d_in[4];
  const float* pconv1_w = (const float*)d_in[5];
  const float* pconv1_b = (const float*)d_in[6];
  const float* pconv2_w = (const float*)d_in[7];
  const float* pconv2_b = (const float*)d_in[8];
  const float* conv1_w  = (const float*)d_in[9];
  const float* conv1_b  = (const float*)d_in[10];
  const float* conv2_w  = (const float*)d_in[11];
  const float* conv2_b  = (const float*)d_in[12];
  const float* psconv1_w = (const float*)d_in[13];
  const float* psconv1_b = (const float*)d_in[14];
  const float* psconv2_w = (const float*)d_in[15];
  const float* psconv2_b = (const float*)d_in[16];
  const float* final_w  = (const float*)d_in[17];
  const float* final_b  = (const float*)d_in[18];
  const float* seg1_w = (const float*)d_in[19];
  const float* seg1_b = (const float*)d_in[20];
  const float* seg2_w = (const float*)d_in[21];
  const float* seg2_b = (const float*)d_in[22];
  const float* seg3_w = (const float*)d_in[23];
  const float* seg3_b = (const float*)d_in[24];
  const float* dis1_w = (const float*)d_in[25];
  const float* dis1_b = (const float*)d_in[26];
  const float* dis2_w = (const float*)d_in[27];
  const float* dis2_b = (const float*)d_in[28];
  const float* dis3_w = (const float*)d_in[29];
  const float* dis3_b = (const float*)d_in[30];
  const float* lin1_w = (const float*)d_in[31];
  const float* lin1_b = (const float*)d_in[32];
  const float* lin2_w = (const float*)d_in[33];
  const float* lin2_b = (const float*)d_in[34];
  const float* lin3_w = (const float*)d_in[35];
  const float* lin3_b = (const float*)d_in[36];
  float* out = (float*)d_out;

  char* ws = (char*)d_ws;
  size_t off = 0;
  auto carve = [&](size_t bytes) -> void* {
    void* p = ws + off;
    off = (off + bytes + 255) & ~(size_t)255;
    return p;
  };
  _Float16* feat_h    = (_Float16*)carve((size_t)NUM_PT * 32 * 2);
  _Float16* feat_lr   = (_Float16*)carve((size_t)NUM_PT * 32 * 2);
  _Float16* pf_h      = (_Float16*)carve((size_t)NUM_PT * 128 * 2);
  _Float16* pf_lr     = (_Float16*)carve((size_t)NUM_PT * 128 * 2);
  int*      idxArr    = (int*)carve((size_t)NUM_PT * TOPK * 4);
  float*    wgtArr    = (float*)carve((size_t)NUM_PT * TOPK * 4);
  _Float16* fpool_lr  = (_Float16*)carve((size_t)NUM_PT * 128 * 2);
  _Float16* fppool_lr = (_Float16*)carve((size_t)NUM_PT * 128 * 2);
  float*    fused     = (float*)carve((size_t)NUM_PT * 128 * 4);
  _Float16* c1h = (_Float16*)carve(64 * 32 * 2);
  _Float16* c2h = (_Float16*)carve(128 * 64 * 2);
  _Float16* p1h = (_Float16*)carve(256 * 128 * 2);
  _Float16* p2h = (_Float16*)carve(128 * 256 * 2);
  _Float16* fwh = (_Float16*)carve(128 * 416 * 2);
  float*    dcen = (float*)carve(NUM_PT * 4);
  float*    gbuf = (float*)carve(128 * 4);

  // weight conversions
  f2h_kernel<<<(64 * 32 + 255) / 256, 256, 0, stream>>>(conv1_w, c1h, 64 * 32);
  f2h_kernel<<<(128 * 64 + 255) / 256, 256, 0, stream>>>(conv2_w, c2h, 128 * 64);
  f2h_kernel<<<(256 * 128 + 255) / 256, 256, 0, stream>>>(psconv1_w, p1h, 256 * 128);
  f2h_kernel<<<(128 * 256 + 255) / 256, 256, 0, stream>>>(psconv2_w, p2h, 128 * 256);
  f2h_kernel<<<(128 * 416 + 255) / 256, 256, 0, stream>>>(final_w, fwh, 128 * 416);

  gather_feat_kernel<<<NUM_PT / 256, 256, 0, stream>>>(img, choose, feat_h, feat_lr);
  pf_kernel<<<NUM_PT / 256, 256, 0, stream>>>(cloud, pconv1_w, pconv1_b, pconv2_w, pconv2_b,
                                              pf_h, pf_lr);
  knn_kernel<<<NUM_PT / 256, 256, 0, stream>>>(cloud, idxArr, wgtArr);

  point_mlp_kernel<<<NUM_PT / 8, 256, 0, stream>>>(
      feat_h, pf_h, idxArr, wgtArr,
      c1h, conv1_b, c2h, conv2_b,
      p1h, psconv1_b, p2h, psconv2_b,
      fpool_lr, fppool_lr);

  final_gemm_kernel<<<(NUM_PT / 16) / 8, 256, 0, stream>>>(
      fppool_lr, feat_lr, fpool_lr, pf_lr, fwh, final_b, fused);

  heads_kernel<<<NUM_PT / 64, 64, 0, stream>>>(
      fused, seg1_w, seg1_b, seg2_w, seg2_b, seg3_w, seg3_b,
      dis1_w, dis1_b, dis2_w, dis2_b, dis3_w, dis3_b, out);

  discenter_kernel<<<1, 1024, 0, stream>>>(cloud, t, dcen, out);
  gpool_kernel<<<128, 256, 0, stream>>>(fused, dcen, seg, gbuf);
  kp_kernel<<<1, 128, 0, stream>>>(gbuf, lin1_w, lin1_b, lin2_w, lin2_b, lin3_w, lin3_b, out);
}
